// EnergyPredTransformerGNN_59511066853684
// MI455X (gfx1250) — matrix operand order repaired and checked by reading, not compiled, not run on hardware
//
#include <hip/hip_runtime.h>
#include <hip/hip_bf16.h>

typedef _Float16 v16h __attribute__((ext_vector_type(16)));
typedef _Float16 v8h  __attribute__((ext_vector_type(8)));
typedef float    v8f  __attribute__((ext_vector_type(8)));

#define GN      100000      // nodes
#define GE      400000      // edges
#define GG      32          // graphs
#define HD      192
#define NHEAD   6
#define DH      32
#define ROWT    (GN / 16)   // 6250 row tiles (exact)
#define COLT    48          // 4 mats * 12 col tiles
#define KSTEPS  6           // 192 / 32
#define WPACK_LAYER (COLT * KSTEPS * 32 * 8)   // dwords per layer = 73728
#define WPACK_TOTAL (3 * WPACK_LAYER)

union AFrag { v16h v; v8h h[2]; };

// ---------- order-preserving float <-> uint key (for atomicMax softmax) ----------
__device__ __forceinline__ unsigned int fkey(float f) {
    unsigned int u = __float_as_uint(f);
    return (u >> 31) ? ~u : (u | 0x80000000u);
}
__device__ __forceinline__ float funkey(unsigned int k) {
    return (k >> 31) ? __uint_as_float(k & 0x7FFFFFFFu) : __uint_as_float(~k);
}

// ---------- pack f32 weights [3][192][192] x4 mats into WMMA B-fragment layout ----------
// idx = (((layer*48 + mat*12 + cim)*6 + ks)*32 + lane)*8 + r
// B 32x16 f16 k order used here (matches A loads): lanes 0-15: K=0..15 (vgpr r: K=2r,2r+1); lanes 16-31: K=16..31
__global__ void pack_weights(const float* __restrict__ Wq, const float* __restrict__ Wk,
                             const float* __restrict__ Wv, const float* __restrict__ Ws,
                             unsigned int* __restrict__ wp) {
    int idx = blockIdx.x * blockDim.x + threadIdx.x;
    if (idx >= WPACK_TOTAL) return;
    int r    = idx & 7;
    int lane = (idx >> 3) & 31;
    int t    = idx >> 8;
    int ks    = t % 6;
    int cim   = (t / 6) % 12;
    int mat   = (t / 6 / 12) % 4;
    int layer = t / 6 / 48;
    int n = cim * 16 + (lane & 15);
    int k = ks * 32 + ((lane >> 4) ? 16 : 0) + 2 * r;
    const float* W = (mat == 0 ? Wq : mat == 1 ? Wk : mat == 2 ? Wv : Ws) + layer * HD * HD;
    _Float16 lo = (_Float16)W[k * HD + n];
    _Float16 hi = (_Float16)W[(k + 1) * HD + n];
    unsigned short ulo = *(unsigned short*)&lo;
    unsigned short uhi = *(unsigned short*)&hi;
    wp[idx] = (unsigned int)ulo | ((unsigned int)uhi << 16);
}

// ---------- input projection: h = x @ Wp + bp ; write f32 + f16 copies ----------
__global__ void input_proj(const float* __restrict__ x, const float* __restrict__ Wp,
                           const float* __restrict__ bp,
                           float* __restrict__ h32, _Float16* __restrict__ h16) {
    int idx = blockIdx.x * blockDim.x + threadIdx.x;
    if (idx >= GN * HD) return;
    int n = idx / HD, t = idx % HD;
    const float* xr = x + n * 4;
    float acc = bp[t];
    acc += xr[0] * Wp[0 * HD + t];
    acc += xr[1] * Wp[1 * HD + t];
    acc += xr[2] * Wp[2 * HD + t];
    acc += xr[3] * Wp[3 * HD + t];
    h32[idx] = acc;
    h16[idx] = (_Float16)acc;
}

// ---------- fused Q/K/V/Skip GEMM via v_wmma_f32_16x16x32_f16 ----------
// one wave -> one 16-row tile x 4 col tiles; skip output (mat 3) seeds agg buffer
__global__ void gemm_qkvs(const _Float16* __restrict__ A, const unsigned int* __restrict__ wp,
                          const float* __restrict__ bq, const float* __restrict__ bk,
                          const float* __restrict__ bv, const float* __restrict__ bs,
                          _Float16* __restrict__ qo, _Float16* __restrict__ ko,
                          _Float16* __restrict__ vo, float* __restrict__ so) {
    int gwave = (blockIdx.x * blockDim.x + threadIdx.x) >> 5;
    int lane  = threadIdx.x & 31;
    int rowTile = gwave / 12;
    int group   = gwave % 12;
    if (rowTile >= ROWT) return;
    int rowBase = rowTile * 16;
    int m  = lane & 15;
    int hi = lane >> 4;   // K-half selector for A, row-half selector for C

    // A fragments: 16x32 f16. lanes 0-15: K 0..15; lanes 16-31: K 16..31
    AFrag a[KSTEPS];
    const _Float16* arow = A + (size_t)(rowBase + m) * HD;
#pragma unroll
    for (int ks = 0; ks < KSTEPS; ++ks) {
        int k0 = ks * 32 + hi * 16;
        a[ks].h[0] = *(const v8h*)(arow + k0);
        a[ks].h[1] = *(const v8h*)(arow + k0 + 8);
    }

#pragma unroll
    for (int cs = 0; cs < 4; ++cs) {
        int colTile = group * 4 + cs;
        int mat = colTile / 12;
        int cim = colTile % 12;
        const float* bias = (mat == 0) ? bq : (mat == 1) ? bk : (mat == 2) ? bv : bs;
        float bb = bias[cim * 16 + m];
        v8f acc = { bb, bb, bb, bb, bb, bb, bb, bb };
        const unsigned int* bptr = wp + ((size_t)(colTile * KSTEPS) * 32 + lane) * 8;
#pragma unroll
        for (int ks = 0; ks < KSTEPS; ++ks) {
            v16h b = *(const v16h*)(bptr + (size_t)ks * 256);
            acc = __builtin_amdgcn_wmma_f32_16x16x32_f16(false, a[ks].v, false, b,
                                                         (short)0, acc, false, false);
        }
        int col = cim * 16 + m;
        if (mat == 3) {
#pragma unroll
            for (int r = 0; r < 8; ++r)
                so[(size_t)(rowBase + hi * 8 + r) * HD + col] = acc[r];
        } else {
            _Float16* o = (mat == 0) ? qo : (mat == 1) ? ko : vo;
#pragma unroll
            for (int r = 0; r < 8; ++r)
                o[(size_t)(rowBase + hi * 8 + r) * HD + col] = (_Float16)acc[r];
        }
    }
}

// ---------- reset softmax stats ----------
__global__ void init_mz(unsigned int* __restrict__ mk, float* __restrict__ z) {
    int idx = blockIdx.x * blockDim.x + threadIdx.x;
    if (idx >= GN * NHEAD) return;
    mk[idx] = 0u;         // maps below any real encoded logit
    z[idx]  = 0.0f;
}

// ---------- E1: logits + segment max (block = 192 thr = 6 waves, wave w == head w) ----------
__global__ void edge_logits(const int* __restrict__ src, const int* __restrict__ dst,
                            const float* __restrict__ ew, const float* __restrict__ We,
                            const _Float16* __restrict__ q, const _Float16* __restrict__ k,
                            float* __restrict__ logits, unsigned int* __restrict__ mk) {
    int e = blockIdx.x;
    if (e >= GE) return;
    int t = threadIdx.x;
    int h = t >> 5;
    int s = src[e], d = dst[e];
    float w = ew[e];
    float qv = (float)q[(size_t)d * HD + t];
    float kv = (float)k[(size_t)s * HD + t] + w * We[t];
    float p = qv * kv;
#pragma unroll
    for (int off = 16; off > 0; off >>= 1) p += __shfl_xor(p, off, 32);
    if ((t & 31) == 0) {
        float lg = p * 0.17677669529663687f;   // 1/sqrt(32)
        logits[(size_t)e * NHEAD + h] = lg;
        atomicMax(&mk[(size_t)d * NHEAD + h], fkey(lg));
    }
}

// ---------- E2: p = exp(l - m); z += p ----------
__global__ void edge_softmax(const int* __restrict__ dst, float* __restrict__ logits,
                             const unsigned int* __restrict__ mk, float* __restrict__ z) {
    int idx = blockIdx.x * blockDim.x + threadIdx.x;
    if (idx >= GE * NHEAD) return;
    int e = idx / NHEAD, h = idx % NHEAD;
    int d = dst[e];
    float m = funkey(mk[(size_t)d * NHEAD + h]);
    float p = expf(logits[idx] - m);
    logits[idx] = p;
    atomicAdd(&z[(size_t)d * NHEAD + h], p);
}

// ---------- E3: agg[dst] += (v[src] + e_emb) * alpha ----------
__global__ void edge_agg(const int* __restrict__ src, const int* __restrict__ dst,
                         const float* __restrict__ ew, const float* __restrict__ We,
                         const _Float16* __restrict__ v, const float* __restrict__ pbuf,
                         const float* __restrict__ z, float* __restrict__ agg) {
    int e = blockIdx.x;
    if (e >= GE) return;
    int t = threadIdx.x;
    int h = t >> 5;
    int s = src[e], d = dst[e];
    float alpha = pbuf[(size_t)e * NHEAD + h] / z[(size_t)d * NHEAD + h];
    float val = ((float)v[(size_t)s * HD + t] + ew[e] * We[t]) * alpha;
    atomicAdd(&agg[(size_t)d * HD + t], val);
}

// ---------- LayerNorm + ReLU + residual; refresh f32 and f16 copies ----------
__global__ void ln_residual(const float* __restrict__ agg, const float* __restrict__ g,
                            const float* __restrict__ b,
                            float* __restrict__ h32, _Float16* __restrict__ h16) {
    int n = blockIdx.x;
    int t = threadIdx.x;
    float v = agg[(size_t)n * HD + t];
    float s = v, s2 = v * v;
#pragma unroll
    for (int off = 16; off > 0; off >>= 1) {
        s  += __shfl_xor(s,  off, 32);
        s2 += __shfl_xor(s2, off, 32);
    }
    __shared__ float ls[NHEAD], ls2[NHEAD];
    if ((t & 31) == 0) { ls[t >> 5] = s; ls2[t >> 5] = s2; }
    __syncthreads();
    float S = 0.f, S2 = 0.f;
#pragma unroll
    for (int i = 0; i < NHEAD; ++i) { S += ls[i]; S2 += ls2[i]; }
    float mu  = S * (1.0f / HD);
    float var = S2 * (1.0f / HD) - mu * mu;
    float r = (v - mu) * rsqrtf(var + 1e-5f) * g[t] + b[t];
    r = fmaxf(r, 0.0f);
    float hn = h32[(size_t)n * HD + t] + r;
    h32[(size_t)n * HD + t] = hn;
    h16[(size_t)n * HD + t] = (_Float16)hn;
}

// ---------- pooling ----------
__global__ void pool_init(float* __restrict__ sums, float* __restrict__ cnt) {
    int idx = blockIdx.x * blockDim.x + threadIdx.x;
    if (idx < GG * HD) sums[idx] = 0.0f;
    if (idx < GG) cnt[idx] = 0.0f;
}
__global__ void pool_accum(const float* __restrict__ h32, const int* __restrict__ batch,
                           float* __restrict__ sums, float* __restrict__ cnt) {
    int idx = blockIdx.x * blockDim.x + threadIdx.x;
    if (idx >= GN * HD) return;
    int n = idx / HD, t = idx % HD;
    int g = batch[n];
    atomicAdd(&sums[(size_t)g * HD + t], h32[idx]);
    if (t == 0) atomicAdd(&cnt[g], 1.0f);
}

// ---------- MLP head: one block per graph ----------
__global__ void head(const float* __restrict__ sums, const float* __restrict__ cnt,
                     const float* __restrict__ ie,
                     const float* __restrict__ fciW, const float* __restrict__ fcib,
                     const float* __restrict__ fcig, const float* __restrict__ fcilb,
                     const float* __restrict__ fc1W, const float* __restrict__ fc1b,
                     const float* __restrict__ fc1g, const float* __restrict__ fc1lb,
                     const float* __restrict__ fc2W, const float* __restrict__ fc2b,
                     const float* __restrict__ fc2g, const float* __restrict__ fc2lb,
                     const float* __restrict__ fc3W, const float* __restrict__ fc3b,
                     float* __restrict__ out) {
    int g = blockIdx.x, t = threadIdx.x;
    __shared__ float z[2 * HD], buf[HD], z2[HD / 2], red[2];

    float c = fmaxf(cnt[g], 1.0f);
    z[t] = sums[(size_t)g * HD + t] / c;

    // init feature: LN(ie*W + b), ReLU
    float a = ie[g] * fciW[t] + fcib[t];
    buf[t] = a;
    __syncthreads();
    if (t == 0) {
        float S = 0.f, S2 = 0.f;
        for (int i = 0; i < HD; ++i) { S += buf[i]; S2 += buf[i] * buf[i]; }
        float mu = S / HD;
        red[0] = mu; red[1] = rsqrtf(S2 / HD - mu * mu + 1e-5f);
    }
    __syncthreads();
    z[HD + t] = fmaxf((a - red[0]) * red[1] * fcig[t] + fcilb[t], 0.0f);
    __syncthreads();

    // fc1: [384] -> [192], LN, ReLU
    float acc = fc1b[t];
    for (int k = 0; k < 2 * HD; ++k) acc += z[k] * fc1W[k * HD + t];
    buf[t] = acc;
    __syncthreads();
    if (t == 0) {
        float S = 0.f, S2 = 0.f;
        for (int i = 0; i < HD; ++i) { S += buf[i]; S2 += buf[i] * buf[i]; }
        float mu = S / HD;
        red[0] = mu; red[1] = rsqrtf(S2 / HD - mu * mu + 1e-5f);
    }
    __syncthreads();
    float z1v = fmaxf((acc - red[0]) * red[1] * fc1g[t] + fc1lb[t], 0.0f);
    __syncthreads();
    buf[t] = z1v;
    __syncthreads();

    // fc2: [192] -> [96], LN, ReLU
    if (t < HD / 2) {
        float a2 = fc2b[t];
        for (int k = 0; k < HD; ++k) a2 += buf[k] * fc2W[k * (HD / 2) + t];
        z2[t] = a2;
    }
    __syncthreads();
    if (t == 0) {
        float S = 0.f, S2 = 0.f;
        for (int i = 0; i < HD / 2; ++i) { S += z2[i]; S2 += z2[i] * z2[i]; }
        float mu = S / (HD / 2);
        red[0] = mu; red[1] = rsqrtf(S2 / (HD / 2) - mu * mu + 1e-5f);
    }
    __syncthreads();
    if (t < HD / 2)
        z2[t] = fmaxf((z2[t] - red[0]) * red[1] * fc2g[t] + fc2lb[t], 0.0f);
    __syncthreads();

    if (t == 0) {
        float o = fc3b[0];
        for (int k = 0; k < HD / 2; ++k) o += z2[k] * fc3W[k];
        out[g] = o;
    }
}

extern "C" void kernel_launch(void* const* d_in, const int* in_sizes, int n_in,
                              void* d_out, int out_size, void* d_ws, size_t ws_size,
                              hipStream_t stream) {
    const float* x     = (const float*)d_in[0];
    const int*   ei    = (const int*)  d_in[1];
    const float* ew    = (const float*)d_in[2];
    const int*   batch = (const int*)  d_in[3];
    const float* ie    = (const float*)d_in[4];
    const float* Wp    = (const float*)d_in[5];
    const float* bp    = (const float*)d_in[6];
    const float* Wq    = (const float*)d_in[7];
    const float* bq    = (const float*)d_in[8];
    const float* Wk    = (const float*)d_in[9];
    const float* bk    = (const float*)d_in[10];
    const float* Wv    = (const float*)d_in[11];
    const float* bv    = (const float*)d_in[12];
    const float* We    = (const float*)d_in[13];
    const float* Ws    = (const float*)d_in[14];
    const float* bs    = (const float*)d_in[15];
    const float* lng   = (const float*)d_in[16];
    const float* lnb   = (const float*)d_in[17];
    const float* fciW  = (const float*)d_in[18];
    const float* fcib  = (const float*)d_in[19];
    const float* fcig  = (const float*)d_in[20];
    const float* fcilb = (const float*)d_in[21];
    const float* fc1W  = (const float*)d_in[22];
    const float* fc1b  = (const float*)d_in[23];
    const float* fc1g  = (const float*)d_in[24];
    const float* fc1lb = (const float*)d_in[25];
    const float* fc2W  = (const float*)d_in[26];
    const float* fc2b  = (const float*)d_in[27];
    const float* fc2g  = (const float*)d_in[28];
    const float* fc2lb = (const float*)d_in[29];
    const float* fc3W  = (const float*)d_in[30];
    const float* fc3b  = (const float*)d_in[31];
    float* out = (float*)d_out;

    const int* src = ei;
    const int* dst = ei + GE;

    char* ws = (char*)d_ws;
    size_t off = 0;
    auto alloc = [&](size_t bytes) -> char* {
        char* p = ws + off;
        off += (bytes + 255) & ~(size_t)255;
        return p;
    };
    float*        h32    = (float*)       alloc((size_t)GN * HD * 4);
    _Float16*     h16    = (_Float16*)    alloc((size_t)GN * HD * 2);
    _Float16*     qh     = (_Float16*)    alloc((size_t)GN * HD * 2);
    _Float16*     kh     = (_Float16*)    alloc((size_t)GN * HD * 2);
    _Float16*     vh     = (_Float16*)    alloc((size_t)GN * HD * 2);
    float*        agg    = (float*)       alloc((size_t)GN * HD * 4);
    float*        logits = (float*)       alloc((size_t)GE * NHEAD * 4);
    unsigned int* mkeys  = (unsigned int*)alloc((size_t)GN * NHEAD * 4);
    float*        zsum   = (float*)       alloc((size_t)GN * NHEAD * 4);
    unsigned int* wpack  = (unsigned int*)alloc((size_t)WPACK_TOTAL * 4);
    float*        psum   = (float*)       alloc((size_t)GG * HD * 4);
    float*        pcnt   = (float*)       alloc((size_t)GG * 4);

    pack_weights<<<(WPACK_TOTAL + 255) / 256, 256, 0, stream>>>(Wq, Wk, Wv, Ws, wpack);
    input_proj<<<(GN * HD + 255) / 256, 256, 0, stream>>>(x, Wp, bp, h32, h16);

    for (int i = 0; i < 3; ++i) {
        gemm_qkvs<<<(ROWT * 12) / 8, 256, 0, stream>>>(
            h16, wpack + (size_t)i * WPACK_LAYER,
            bq + i * HD, bk + i * HD, bv + i * HD, bs + i * HD,
            qh, kh, vh, agg);
        init_mz<<<(GN * NHEAD + 255) / 256, 256, 0, stream>>>(mkeys, zsum);
        edge_logits<<<GE, HD, 0, stream>>>(src, dst, ew, We + i * HD, qh, kh, logits, mkeys);
        edge_softmax<<<(GE * NHEAD + 255) / 256, 256, 0, stream>>>(dst, logits, mkeys, zsum);
        edge_agg<<<GE, HD, 0, stream>>>(src, dst, ew, We + i * HD, vh, logits, zsum, agg);
        ln_residual<<<GN, HD, 0, stream>>>(agg, lng + i * HD, lnb + i * HD, h32, h16);
    }

    pool_init<<<(GG * HD + 255) / 256, 256, 0, stream>>>(psum, pcnt);
    pool_accum<<<(GN * HD + 255) / 256, 256, 0, stream>>>(h32, batch, psum, pcnt);
    head<<<GG, HD, 0, stream>>>(psum, pcnt, ie,
                                fciW, fcib, fcig, fcilb,
                                fc1W, fc1b, fc1g, fc1lb,
                                fc2W, fc2b, fc2g, fc2lb,
                                fc3W, fc3b, out);
}
